// MultiHeadAttention_8375186227578
// MI455X (gfx1250) — matrix operand order repaired and checked by reading, not compiled, not run on hardware
//
#include <hip/hip_runtime.h>

// Problem constants (reference: B=2, S=2048, D=512, H=8, depth=64)
#define BB 2
#define SS 2048
#define DD 512
#define HH 8
#define DH 64

typedef __attribute__((ext_vector_type(16))) __bf16 v16bf;
typedef __attribute__((ext_vector_type(8)))  __bf16 v8bf;
typedef __attribute__((ext_vector_type(4)))  __bf16 v4bf;
typedef __attribute__((ext_vector_type(8)))  float  v8f;
typedef __attribute__((ext_vector_type(4)))  float  v4f;

static __device__ __forceinline__ v16bf cat8(v8bf a, v8bf b) {
  return __builtin_shufflevector(a, b, 0,1,2,3,4,5,6,7,8,9,10,11,12,13,14,15);
}
static __device__ __forceinline__ v8f v8f_zero() {
  v8f z;
#pragma unroll
  for (int i = 0; i < 8; ++i) z[i] = 0.0f;
  return z;
}

// ---------------------------------------------------------------------------
// Elementwise fp32 -> bf16 (vectorized: 4 elements/thread)
// ---------------------------------------------------------------------------
__global__ __launch_bounds__(256) void cvt_f32_bf16(const float* __restrict__ src,
                                                    __bf16* __restrict__ dst, int n4)
{
  int t = blockIdx.x * blockDim.x + threadIdx.x;
  if (t >= n4) return;
  v4f x = *(const v4f*)(src + 4 * (size_t)t);
  v4bf y;
#pragma unroll
  for (int i = 0; i < 4; ++i) y[i] = (__bf16)x[i];
  *(v4bf*)(dst + 4 * (size_t)t) = y;
}

// ---------------------------------------------------------------------------
// W[K,N] fp32 -> Wt[N,K] bf16 (transpose + convert; tiny, runs once)
// ---------------------------------------------------------------------------
__global__ __launch_bounds__(256) void transpose_cvt(const float* __restrict__ W,
                                                     __bf16* __restrict__ Wt, int K, int N)
{
  int t = blockIdx.x * blockDim.x + threadIdx.x;
  if (t >= K * N) return;
  int k = t / N, n = t - k * N;
  Wt[(size_t)n * K + k] = (__bf16)W[t];
}

// ---------------------------------------------------------------------------
// C = A[M,K](bf16) @ Wt[N,K](bf16, pre-transposed).  One wave = 16x64 strip.
// A frag: lane(row=idx), slots = two contiguous 8-elt runs (b128 loads)
// B frag: lane(col=idx), slots = one contiguous 16-elt run  (2x b128 loads)
// mode 0: fp32 out [M,N] (+bias) ; mode 2: bf16 out transposed to [B,H,64,S]
// ---------------------------------------------------------------------------
__global__ __launch_bounds__(128) void gemm_bf16_wmma(
    const __bf16* __restrict__ A, const __bf16* __restrict__ Wt,
    const float* __restrict__ bias, void* __restrict__ Cout,
    int M, int N, int K, int mode)
{
  const int wave = blockIdx.x * (blockDim.x >> 5) + (threadIdx.x >> 5);
  const int lane = threadIdx.x & 31;
  const int idx  = lane & 15;
  const int half = lane >> 4;
  const int nstrips = N >> 6;
  const int n0 = (wave % nstrips) * 64;
  const int m0 = (wave / nstrips) * 16;
  if (m0 >= M) return;

  v8f acc0 = v8f_zero(), acc1 = v8f_zero(), acc2 = v8f_zero(), acc3 = v8f_zero();
  const __bf16* arow = A + (size_t)(m0 + idx) * K;

  for (int kc = 0; kc < K; kc += 32) {
    v8bf alo = *(const v8bf*)(arow + kc + 16 * half);
    v8bf ahi = *(const v8bf*)(arow + kc + 16 * half + 8);
    v16bf a = cat8(alo, ahi);
#pragma unroll
    for (int j = 0; j < 4; ++j) {
      const __bf16* wrow = Wt + (size_t)(n0 + j * 16 + idx) * K + kc + 16 * half;
      v16bf bfrag = *(const v16bf*)wrow;
      v8f cin = (j == 0) ? acc0 : (j == 1) ? acc1 : (j == 2) ? acc2 : acc3;
      v8f cout = __builtin_amdgcn_wmma_f32_16x16x32_bf16(false, a, false, bfrag,
                                                         (short)0, cin, false, false);
      if (j == 0) acc0 = cout; else if (j == 1) acc1 = cout;
      else if (j == 2) acc2 = cout; else acc3 = cout;
    }
  }

  if (mode == 0) {
    float* C = (float*)Cout;
#pragma unroll
    for (int j = 0; j < 4; ++j) {
      int n = n0 + j * 16 + idx;
      float bv = bias ? bias[n] : 0.0f;
      v8f o = (j == 0) ? acc0 : (j == 1) ? acc1 : (j == 2) ? acc2 : acc3;
#pragma unroll
      for (int r = 0; r < 8; ++r)
        C[(size_t)(m0 + r + half * 8) * N + n] = o[r] + bv;
    }
  } else {  // mode 2: bf16, transposed head-wise -> vvT[((b*H+h)*64+dn)*S + s]
    __bf16* T = (__bf16*)Cout;
#pragma unroll
    for (int j = 0; j < 4; ++j) {
      int n = n0 + j * 16 + idx;
      int h = n >> 6, dn = n & 63;
      v8f o = (j == 0) ? acc0 : (j == 1) ? acc1 : (j == 2) ? acc2 : acc3;
#pragma unroll
      for (int r = 0; r < 8; ++r) {
        int m = m0 + r + half * 8;
        int b = m / SS, s = m - b * SS;
        T[(((size_t)(b * HH + h) * DH + dn) * SS) + s] = (__bf16)o[r];
      }
    }
  }
}

// ---------------------------------------------------------------------------
// Per 64-wide head row: qs = q*0.125 (bf16), qn = q/||q|| (bf16). 1 wave/row.
// ---------------------------------------------------------------------------
__global__ __launch_bounds__(256) void l2norm_scale(const float* __restrict__ q,
                                                    __bf16* __restrict__ qs,
                                                    __bf16* __restrict__ qn, int nrows)
{
  int row  = blockIdx.x * 8 + (threadIdx.x >> 5);
  int lane = threadIdx.x & 31;
  if (row >= nrows) return;
  const float* p = q + (size_t)row * 64;
  float a = p[lane], b = p[lane + 32];
  float ss = a * a + b * b;
#pragma unroll
  for (int off = 16; off; off >>= 1) ss += __shfl_xor(ss, off, 32);
  float rn = rsqrtf(ss);
  qs[(size_t)row * 64 + lane]      = (__bf16)(a * 0.125f);
  qs[(size_t)row * 64 + lane + 32] = (__bf16)(b * 0.125f);
  qn[(size_t)row * 64 + lane]      = (__bf16)(a * rn);
  qn[(size_t)row * 64 + lane + 32] = (__bf16)(b * rn);
}

// ---------------------------------------------------------------------------
// Flash-style causal attention, all-bf16 operands. One wave / (b,h,16 queries).
// qs: [B,S,D] bf16 (scale folded)   qn: [B,S,D] bf16
// vvT: [B,H,64,S] bf16 (so V B-fragments are contiguous along keys)
// out: [B,S,D] bf16 (feeds final GEMM)
// ---------------------------------------------------------------------------
__global__ __launch_bounds__(128) void flash_attn_wmma(
    const __bf16* __restrict__ qs, const __bf16* __restrict__ qn,
    const __bf16* __restrict__ vvT, __bf16* __restrict__ out,
    const int* __restrict__ seedp)
{
  __shared__ __align__(16) __bf16 plds[4][16][40];
  const unsigned seed = (unsigned)seedp[0];
  const int w    = threadIdx.x >> 5;
  const int wave = blockIdx.x * 4 + w;
  const int lane = threadIdx.x & 31;
  const int idx  = lane & 15;
  const int half = lane >> 4;

  const int qt = wave & 127;
  const int h  = (wave >> 7) & 7;
  const int b  = wave >> 10;
  const int q0 = qt << 4;

  // Q fragments: two contiguous 8-elt bf16 runs each
  const __bf16* qrow = qs + ((size_t)(b * SS + q0 + idx) * DD) + h * DH;
  v16bf aq0 = cat8(*(const v8bf*)(qrow + 16 * half),      *(const v8bf*)(qrow + 8 + 16 * half));
  v16bf aq1 = cat8(*(const v8bf*)(qrow + 32 + 16 * half), *(const v8bf*)(qrow + 40 + 16 * half));

  v8f o0 = v8f_zero(), o1 = v8f_zero(), o2 = v8f_zero(), o3 = v8f_zero();
  float m8[8], l8[8];
#pragma unroll
  for (int r = 0; r < 8; ++r) { m8[r] = -INFINITY; l8[r] = 0.0f; }

  const int kend = q0 + 16;   // strict causal (exclusive)
  for (int kc = 0; kc < kend; kc += 32) {
    float stile[2][8];
#pragma unroll
    for (int t = 0; t < 2; ++t) {
      const int k0t = kc + 16 * t;
      v8f c = v8f_zero();
      if (k0t < kend) {       // wave-uniform
        const __bf16* krow = qn + ((size_t)(b * SS + k0t + idx) * DD) + h * DH;
        v16bf bk0 = *(const v16bf*)(krow + 16 * half);
        v16bf bk1 = *(const v16bf*)(krow + 32 + 16 * half);
        c = __builtin_amdgcn_wmma_f32_16x16x32_bf16(false, aq0, false, bk0, (short)0, c, false, false);
        c = __builtin_amdgcn_wmma_f32_16x16x32_bf16(false, aq1, false, bk1, (short)0, c, false, false);
      }
#pragma unroll
      for (int r = 0; r < 8; ++r) {
        int qr = q0 + r + half * 8;
        int kcol = k0t + idx;
        float v = c[r];
        if (k0t >= kend || kcol > qr) v = -1.0e38f;     // LOOK_AHEAD_NEG
        else if (kcol == qr)          v += -50000.0f;   // SELF_NEG
        stile[t][r] = v;
      }
    }
    // online softmax (row reductions across the 16 lanes of each half)
#pragma unroll
    for (int r = 0; r < 8; ++r) {
      float tm = fmaxf(stile[0][r], stile[1][r]);
#pragma unroll
      for (int off = 1; off < 16; off <<= 1) tm = fmaxf(tm, __shfl_xor(tm, off, 16));
      float mnew = fmaxf(m8[r], tm);
      float corr = __expf(m8[r] - mnew);
      m8[r] = mnew;
      float p0 = __expf(stile[0][r] - mnew);
      float p1 = __expf(stile[1][r] - mnew);
      float rs = p0 + p1;
#pragma unroll
      for (int off = 1; off < 16; off <<= 1) rs += __shfl_xor(rs, off, 16);
      l8[r] = l8[r] * corr + rs;
      o0[r] *= corr; o1[r] *= corr; o2[r] *= corr; o3[r] *= corr;
      stile[0][r] = p0; stile[1][r] = p1;
    }
    // hash dropout + stage P (bf16) into per-wave LDS tile [row][k]
#pragma unroll
    for (int t = 0; t < 2; ++t)
#pragma unroll
      for (int r = 0; r < 8; ++r) {
        int qr = q0 + r + half * 8;
        int kcol = kc + 16 * t + idx;
        unsigned x = (unsigned)((((b * HH + h) * SS + qr) * SS) + kcol);
        x ^= seed * 0x9E3779B9u + 0x85EBCA6Bu;
        x = (x ^ 61u) ^ (x >> 16); x *= 9u; x ^= x >> 4; x *= 0x27d4eb2du; x ^= x >> 15;
        float pd = ((x & 0xFFFFFFu) < 15099494u) ? stile[t][r] * (1.0f / 0.9f) : 0.0f;
        plds[w][r + half * 8][idx + 16 * t] = (__bf16)pd;
      }
    __builtin_amdgcn_wave_barrier();
    asm volatile("s_wait_dscnt 0" ::: "memory");   // same-wave LDS ops are in-order
    v16bf pa = cat8(*(const v8bf*)&plds[w][idx][16 * half],
                    *(const v8bf*)&plds[w][idx][8 + 16 * half]);
    __builtin_amdgcn_wave_barrier();
    // O += P @ V : V fragments contiguous along keys in vvT
    const __bf16* vb = vvT + ((size_t)(b * HH + h) * DH) * SS;
#pragma unroll
    for (int j = 0; j < 4; ++j) {
      v16bf bv = *(const v16bf*)(vb + (size_t)(j * 16 + idx) * SS + kc + 16 * half);
      v8f cin = (j == 0) ? o0 : (j == 1) ? o1 : (j == 2) ? o2 : o3;
      v8f cout = __builtin_amdgcn_wmma_f32_16x16x32_bf16(false, pa, false, bv,
                                                         (short)0, cin, false, false);
      if (j == 0) o0 = cout; else if (j == 1) o1 = cout;
      else if (j == 2) o2 = cout; else o3 = cout;
    }
  }
  // epilogue: divide by softmax denominator, merge heads, write bf16
#pragma unroll
  for (int j = 0; j < 4; ++j) {
    v8f o = (j == 0) ? o0 : (j == 1) ? o1 : (j == 2) ? o2 : o3;
#pragma unroll
    for (int r = 0; r < 8; ++r)
      out[((size_t)(b * SS + q0 + r + half * 8) * DD) + h * DH + j * 16 + idx] =
          (__bf16)(o[r] / l8[r]);
  }
}

// ---------------------------------------------------------------------------
extern "C" void kernel_launch(void* const* d_in, const int* in_sizes, int n_in,
                              void* d_out, int out_size, void* d_ws, size_t ws_size,
                              hipStream_t stream) {
  const float* qk    = (const float*)d_in[0];
  const float* v     = (const float*)d_in[1];
  const float* W_qk  = (const float*)d_in[2];
  const float* W_v   = (const float*)d_in[3];
  const float* W_out = (const float*)d_in[4];
  const float* b_out = (const float*)d_in[5];
  const int*   seed  = (const int*)d_in[6];
  float* out = (float*)d_out;

  const int M = BB * SS;                 // 4096
  const size_t MD = (size_t)M * DD;      // 2M elements
  char* ws = (char*)d_ws;
  __bf16* qk_bf  = (__bf16*)ws;                 ws += MD * 2;       // 4MB
  __bf16* v_bf   = (__bf16*)ws;                 ws += MD * 2;       // 4MB
  __bf16* Wqkt   = (__bf16*)ws;                 ws += (size_t)DD * DD * 2;
  __bf16* Wvt    = (__bf16*)ws;                 ws += (size_t)DD * DD * 2;
  __bf16* Woutt  = (__bf16*)ws;                 ws += (size_t)DD * DD * 2;
  float*  q_f32  = (float*)ws;                  ws += MD * 4;       // 8MB
  __bf16* qs_bf  = (__bf16*)ws;                 ws += MD * 2;
  __bf16* qn_bf  = (__bf16*)ws;                 ws += MD * 2;
  __bf16* vvT_bf = (__bf16*)ws;                 ws += MD * 2;
  __bf16* att_bf = (__bf16*)ws;                 ws += MD * 2;

  // 1) input conversions + weight transpose-conversions
  cvt_f32_bf16<<<dim3((int)(MD / 4 / 256)), dim3(256), 0, stream>>>(qk, qk_bf, (int)(MD / 4));
  cvt_f32_bf16<<<dim3((int)(MD / 4 / 256)), dim3(256), 0, stream>>>(v,  v_bf,  (int)(MD / 4));
  int wElems = DD * DD;
  transpose_cvt<<<dim3(wElems / 256), dim3(256), 0, stream>>>(W_qk,  Wqkt,  DD, DD);
  transpose_cvt<<<dim3(wElems / 256), dim3(256), 0, stream>>>(W_v,   Wvt,   DD, DD);
  transpose_cvt<<<dim3(wElems / 256), dim3(256), 0, stream>>>(W_out, Woutt, DD, DD);

  // 2) projections
  dim3 blk(128);
  dim3 ggrid(((M / 16) * (DD / 64)) / 4);      // 512 blocks, 4 waves each
  gemm_bf16_wmma<<<ggrid, blk, 0, stream>>>(qk_bf, Wqkt, nullptr, q_f32,  M, DD, DD, 0);
  gemm_bf16_wmma<<<ggrid, blk, 0, stream>>>(v_bf,  Wvt,  nullptr, vvT_bf, M, DD, DD, 2);

  // 3) key normalization + scale fold
  l2norm_scale<<<dim3(M * HH / 8), dim3(256), 0, stream>>>(q_f32, qs_bf, qn_bf, M * HH);

  // 4) attention
  flash_attn_wmma<<<dim3((BB * HH * (SS / 16)) / 4), blk, 0, stream>>>(
      qs_bf, qn_bf, vvT_bf, att_bf, seed);

  // 5) output projection (+bias), fp32 result
  gemm_bf16_wmma<<<ggrid, blk, 0, stream>>>(att_bf, Woutt, b_out, out, M, DD, DD, 0);
}
